// MultiheadAttention_17789754540085
// MI455X (gfx1250) — matrix operand order repaired and checked by reading, not compiled, not run on hardware
//
#include <hip/hip_runtime.h>
#include <hip/hip_bf16.h>

#define N_NODE   512
#define N_GRAPH  16
#define EMBED    768
#define NHEADS   12
#define HDIM     64
#define NROWS    (N_NODE * N_GRAPH)   // 8192
#define NB       (N_GRAPH * NHEADS)   // 192 head-batches
#define NCH      (EMBED / 32)         // 24 k-chunks per row
#define S_LD     68                   // floats; rows stay 16B aligned

typedef __attribute__((ext_vector_type(16))) _Float16 v16h;
typedef __attribute__((ext_vector_type(8)))  float    v8f;
typedef __attribute__((ext_vector_type(4)))  unsigned v4u;
typedef __attribute__((ext_vector_type(8)))  int      v8i;
typedef __attribute__((ext_vector_type(4)))  int      v4i;

#define HAS_TDM __has_builtin(__builtin_amdgcn_tensor_load_to_lds)

__device__ __forceinline__ v8f wmma16(v16h a, v16h b, v8f c) {
  return __builtin_amdgcn_wmma_f32_16x16x32_f16(
      false, a, false, b, (short)0, c, false, false);
}

// Fragment regions: 32 lanes x 16 contiguous halves (32B/lane) in LDS.
__device__ __forceinline__ v16h load_frag(const _Float16* region, int lane) {
  return *reinterpret_cast<const v16h*>(region + (lane << 4));
}

__device__ __forceinline__ void store_h2(_Float16* dst, float lo, float hi) {
  union { _Float16 h[2]; unsigned u; } u;
  u.h[0] = (_Float16)lo; u.h[1] = (_Float16)hi;
  *reinterpret_cast<unsigned*>(dst) = u.u;
}

// Inverse WMMA layout maps.
__device__ __forceinline__ int a_lane(int m, int cc) { return m + ((cc >> 4) << 4); }
__device__ __forceinline__ int a_slot(int cc)        { return cc & 15; }
__device__ __forceinline__ int b_lane(int n, int kk) { return n + ((kk >> 4) << 4); }
__device__ __forceinline__ int b_slot(int kk)        { return kk & 15; }

// ---- workspace offset maps (all pre-swizzled fragment layouts) ----
// A-frag matrix ws (x, attn): [m>>6][ch][strip][lane*16+slot]
__device__ __forceinline__ size_t a_ws_off(int m, int e) {
  const int mt = m >> 6, r = m & 63;
  const int strip = r >> 4, m16 = r & 15;
  const int ch = e >> 5, cc = e & 31;
  return ((((size_t)mt * NCH + ch) << 2) + strip) * 512 + (a_lane(m16, cc) << 4) + a_slot(cc);
}
// B-frag weight ws: [n>>6][ch][subtile][lane*16+slot]
__device__ __forceinline__ size_t w_ws_off(int n, int k) {
  const int nt = n >> 6, t = (n & 63) >> 4;
  const int ch = k >> 5, cc = k & 31;
  return ((((size_t)nt * NCH + ch) << 2) + t) * 512 + (b_lane(n & 15, cc) << 4) + b_slot(cc);
}
// Q ws: per (b, qtile): A-frag of (query x d)
__device__ __forceinline__ size_t q_ws_off(int b, int node, int d) {
  const int qt = node >> 6, r = node & 63;
  const int strip = r >> 4, m16 = r & 15;
  const int ch = d >> 5, cc = d & 31;
  return (((size_t)b * 8 + qt) << 12) + (((strip << 1) + ch) << 9) +
         (a_lane(m16, cc) << 4) + a_slot(cc);
}
// K ws: per (b, ktile): B-frag of (d x key)
__device__ __forceinline__ size_t k_ws_off(int b, int node, int d) {
  const int kt = node >> 6, key = node & 63;
  const int t = key >> 4, n16 = key & 15;
  const int ch = d >> 5, cc = d & 31;
  return (((size_t)b * 8 + kt) << 12) + (((t << 1) + ch) << 9) +
         (b_lane(n16, cc) << 4) + b_slot(cc);
}
// V ws: per (b, ktile): B-frag of (key x d)
__device__ __forceinline__ size_t v_ws_off(int b, int node, int d) {
  const int kt = node >> 6, key = node & 63;
  const int t = d >> 4;
  const int ch = key >> 5, cc = key & 31;
  return (((size_t)b * 8 + kt) << 12) + (((t << 1) + ch) << 9) +
         (b_lane(d & 15, cc) << 4) + b_slot(cc);
}

// ---- Tensor Data Mover: contiguous nelem (f16) global -> LDS ----
__device__ __forceinline__ unsigned lds_addr_of(const void* p) {
  // Flat shared address: low 32 bits are the LDS byte address (ISA 10.2).
  return (unsigned)(unsigned long long)p;
}

#if HAS_TDM
__device__ __forceinline__ void tdm_load_lds(void* lds_dst, const void* gsrc,
                                             unsigned nelem) {
  const unsigned long long ga = (unsigned long long)gsrc;
  v4u g0;
  g0[0] = 1u;                                           // count=1, user desc
  g0[1] = lds_addr_of(lds_dst);                         // lds_addr
  g0[2] = (unsigned)ga;                                 // global_addr[31:0]
  g0[3] = (unsigned)((ga >> 32) & 0x01ffffffu) | (2u << 30);  // [56:32] + type=2
  v8i g1;
  g1[0] = 0x00010000;                                   // data_size=1 (2B), mask=0
  g1[1] = (int)(nelem << 16);                           // tensor_dim0[15:0]
  g1[2] = (int)((nelem >> 16) | (1u << 16));            // dim0[31:16], tensor_dim1=1
  g1[3] = (int)(nelem << 16);                           // tile_dim0 = nelem
  g1[4] = 1;                                            // tile_dim1 = 1
  g1[5] = (int)nelem;                                   // tensor_dim0_stride
  g1[6] = 0; g1[7] = 0;
  const v4i z4 = {0, 0, 0, 0};
  const v8i z8 = {0, 0, 0, 0, 0, 0, 0, 0};
  // clang-23 / amdgpu-toolchain 6-arg form (probe-confirmed by round-3 error).
  __builtin_amdgcn_tensor_load_to_lds(g0, g1, z4, z4, z8, 0);
}
#endif

// Cooperative fallback / generic contiguous copy (2 x b128 per thread / 16h).
__device__ __forceinline__ void coop_copy(int tid, int nthreads, _Float16* dst,
                                          const _Float16* src, int nelem) {
  for (int p = tid; p < (nelem >> 4); p += nthreads)
    *reinterpret_cast<v16h*>(dst + ((size_t)p << 4)) =
        *reinterpret_cast<const v16h*>(src + ((size_t)p << 4));
}

// ---------------------------------------------------------------- prep 0
// Swizzle + f32->f16 one weight matrix row per block into B-frag ws.
__global__ __launch_bounds__(256) void wswz_kernel(
    const float* __restrict__ w0, const float* __restrict__ w1,
    const float* __restrict__ w2, const float* __restrict__ w3,
    _Float16* __restrict__ o0, _Float16* __restrict__ o1,
    _Float16* __restrict__ o2, _Float16* __restrict__ o3) {
  const float* src; _Float16* dst;
  switch (blockIdx.y) {
    case 0:  src = w0; dst = o0; break;
    case 1:  src = w1; dst = o1; break;
    case 2:  src = w2; dst = o2; break;
    default: src = w3; dst = o3; break;
  }
  const int n = blockIdx.x;
  for (int p = threadIdx.x; p < EMBED / 2; p += 256) {
    const int k = p << 1;
    const float2 v = *reinterpret_cast<const float2*>(&src[(size_t)n * EMBED + k]);
    store_h2(&dst[w_ws_off(n, k)], v.x, v.y);
  }
}

// Swizzle + convert one x row per block into A-frag ws.
__global__ __launch_bounds__(256) void xswz_kernel(
    const float* __restrict__ x, _Float16* __restrict__ xs) {
  const int m = blockIdx.x;
  for (int p = threadIdx.x; p < EMBED / 2; p += 256) {
    const int k = p << 1;
    const float2 v = *reinterpret_cast<const float2*>(&x[(size_t)m * EMBED + k]);
    store_h2(&xs[a_ws_off(m, k)], v.x, v.y);
  }
}

// ---------------------------------------------------------------- kernel 1
// Fused QKV projection from pre-swizzled x and weights.
__global__ __launch_bounds__(128) void qkv_proj_kernel(
    const _Float16* __restrict__ xs,
    const _Float16* __restrict__ wqs, const _Float16* __restrict__ wks,
    const _Float16* __restrict__ wvs,
    const float* __restrict__ bq, const float* __restrict__ bk,
    const float* __restrict__ bv,
    _Float16* __restrict__ qws, _Float16* __restrict__ kws,
    _Float16* __restrict__ vws) {
  __shared__ __align__(32) _Float16 Af[2048];
  __shared__ __align__(32) _Float16 Bf[3][2048];

  const int tid = threadIdx.x, lane = tid & 31, wave = tid >> 5;
  const int mt = blockIdx.x, nt = blockIdx.y;

  v8f accq[4], acck[4], accv[4];
#pragma unroll
  for (int t = 0; t < 4; ++t)
#pragma unroll
    for (int r = 0; r < 8; ++r) { accq[t][r] = 0.f; acck[t][r] = 0.f; accv[t][r] = 0.f; }

  for (int ch = 0; ch < NCH; ++ch) {
    const size_t abase = ((size_t)mt * NCH + ch) << 11;
    const size_t bbase = ((size_t)nt * NCH + ch) << 11;
#if HAS_TDM
    if (wave == 0) {
      tdm_load_lds(Af,    xs  + abase, 2048);
      tdm_load_lds(Bf[0], wqs + bbase, 2048);
      tdm_load_lds(Bf[1], wks + bbase, 2048);
      tdm_load_lds(Bf[2], wvs + bbase, 2048);
      __builtin_amdgcn_s_wait_tensorcnt(0);
    }
#else
    coop_copy(tid, 128, Af,    xs  + abase, 2048);
    coop_copy(tid, 128, Bf[0], wqs + bbase, 2048);
    coop_copy(tid, 128, Bf[1], wks + bbase, 2048);
    coop_copy(tid, 128, Bf[2], wvs + bbase, 2048);
#endif
    __syncthreads();
    const v16h a = load_frag(&Af[wave << 9], lane);
#pragma unroll
    for (int t = 0; t < 4; ++t) {
      accq[t] = wmma16(a, load_frag(&Bf[0][t << 9], lane), accq[t]);
      acck[t] = wmma16(a, load_frag(&Bf[1][t << 9], lane), acck[t]);
      accv[t] = wmma16(a, load_frag(&Bf[2][t << 9], lane), accv[t]);
    }
    __syncthreads();
  }

  // Add bias (+ q scaling), store pre-swizzled per-head q/k/v.
  const int hh = (lane >> 4) & 1, nc = lane & 15;
#pragma unroll
  for (int t = 0; t < 4; ++t) {
    const int n    = nt * 64 + t * 16 + nc;
    const int head = n >> 6;
    const int d    = n & 63;
    const float bqv = bq[n], bkv = bk[n], bvv = bv[n];
#pragma unroll
    for (int r = 0; r < 8; ++r) {
      const int m    = mt * 64 + wave * 16 + r + 8 * hh;
      const int node = m >> 4;
      const int g    = m & 15;
      const int bb   = g * NHEADS + head;
      qws[q_ws_off(bb, node, d)] = (_Float16)((accq[t][r] + bqv) * 0.125f);
      kws[k_ws_off(bb, node, d)] = (_Float16)(acck[t][r] + bkv);
      vws[v_ws_off(bb, node, d)] = (_Float16)(accv[t][r] + bvv);
    }
  }
}

// ---------------------------------------------------------------- kernel 2
// Flash attention: block = (64 queries x one head); K/V tiles via TDM.
__global__ __launch_bounds__(128) void attention_kernel(
    const _Float16* __restrict__ qws, const _Float16* __restrict__ kws,
    const _Float16* __restrict__ vws, const float* __restrict__ attn_bias,
    const float* __restrict__ attn_mask, const unsigned char* __restrict__ kpm,
    _Float16* __restrict__ aws) {
  __shared__ __align__(32) _Float16 Qf[4096];
  __shared__ __align__(32) _Float16 Kf[4096];
  __shared__ __align__(32) _Float16 Vf[4096];
  __shared__ __align__(32) _Float16 Pf[4096];
  __shared__ __align__(16) float Srow[64 * S_LD];
  __shared__ float mrow[64], lrow[64], scrow[64];

  const int tid = threadIdx.x, lane = tid & 31, wave = tid >> 5;
  const int b = blockIdx.y;
  const int g = b / NHEADS, head = b % NHEADS;
  const int q0 = blockIdx.x * 64;
  const size_t qbase = ((size_t)b * 8 + blockIdx.x) << 12;

#if HAS_TDM
  if (wave == 0) {
    tdm_load_lds(Qf, qws + qbase, 4096);
    __builtin_amdgcn_s_wait_tensorcnt(0);
  }
#else
  coop_copy(tid, 128, Qf, qws + qbase, 4096);
#endif
  if (tid < 64) { mrow[tid] = -3.0e38f; lrow[tid] = 0.f; }

  v8f acc[4];
#pragma unroll
  for (int t = 0; t < 4; ++t)
#pragma unroll
    for (int r = 0; r < 8; ++r) acc[t][r] = 0.f;
  __syncthreads();

  const v16h qa0 = load_frag(&Qf[(wave << 10) + (0 << 9)], lane);
  const v16h qa1 = load_frag(&Qf[(wave << 10) + (1 << 9)], lane);
  const int hh = (lane >> 4) & 1, nc = lane & 15;

  for (int kt = 0; kt < N_NODE / 64; ++kt) {
    const int k0 = kt * 64;
    const size_t kbase = ((size_t)b * 8 + kt) << 12;
#if HAS_TDM
    if (wave == 0) {
      tdm_load_lds(Kf, kws + kbase, 4096);
      tdm_load_lds(Vf, vws + kbase, 4096);
      __builtin_amdgcn_s_wait_tensorcnt(0);
    }
#else
    coop_copy(tid, 128, Kf, kws + kbase, 4096);
    coop_copy(tid, 128, Vf, vws + kbase, 4096);
#endif
    if (kt + 1 < N_NODE / 64) {  // global_prefetch_b8 for next bias tile
      __builtin_prefetch(&attn_bias[((size_t)b * N_NODE + q0 + (tid >> 1)) * N_NODE
                                    + k0 + 64], 0, 1);
    }
    __syncthreads();

    // S = Q * K^T (d contracted, two 32-chunks).
#pragma unroll
    for (int t = 0; t < 4; ++t) {
      v8f s;
#pragma unroll
      for (int r = 0; r < 8; ++r) s[r] = 0.f;
      s = wmma16(qa0, load_frag(&Kf[((t << 1) + 0) << 9], lane), s);
      s = wmma16(qa1, load_frag(&Kf[((t << 1) + 1) << 9], lane), s);
#pragma unroll
      for (int r = 0; r < 8; ++r)
        Srow[(wave * 16 + r + 8 * hh) * S_LD + t * 16 + nc] = s[r];
    }
    __syncthreads();

    // Bias + mask + key padding, float4 wide (dominant HBM stream).
    for (int idx = tid; idx < 64 * 16; idx += 128) {
      const int r  = idx >> 4;
      const int c4 = (idx & 15) << 2;
      const int qg = q0 + r, kg = k0 + c4;
      float4 s = *reinterpret_cast<float4*>(&Srow[r * S_LD + c4]);
      const float4 bsv = *reinterpret_cast<const float4*>(
          &attn_bias[((size_t)b * N_NODE + qg) * N_NODE + kg]);
      const float4 msv = *reinterpret_cast<const float4*>(
          &attn_mask[(size_t)qg * N_NODE + kg]);
      s.x += bsv.x + msv.x; s.y += bsv.y + msv.y;
      s.z += bsv.z + msv.z; s.w += bsv.w + msv.w;
      const unsigned pm = *reinterpret_cast<const unsigned*>(&kpm[g * N_NODE + kg]);
      if (pm) {
        if (pm & 0x000000ffu) s.x = -3.0e38f;
        if (pm & 0x0000ff00u) s.y = -3.0e38f;
        if (pm & 0x00ff0000u) s.z = -3.0e38f;
        if (pm & 0xff000000u) s.w = -3.0e38f;
      }
      *reinterpret_cast<float4*>(&Srow[r * S_LD + c4]) = s;
    }
    __syncthreads();

    // Online softmax per row; write P in A-frag layout.
    if (tid < 64) {
      const float4* rowp = reinterpret_cast<const float4*>(&Srow[tid * S_LD]);
      const float mold = mrow[tid];
      float mt2 = mold;
#pragma unroll
      for (int c = 0; c < 16; ++c) {
        const float4 v = rowp[c];
        mt2 = fmaxf(mt2, fmaxf(fmaxf(v.x, v.y), fmaxf(v.z, v.w)));
      }
      const float sc = __expf(mold - mt2);
      float sum = 0.f;
      const int strip = tid >> 4, m = tid & 15;
#pragma unroll
      for (int c = 0; c < 16; ++c) {
        const float4 v = rowp[c];
        const float p0 = __expf(v.x - mt2), p1 = __expf(v.y - mt2);
        const float p2 = __expf(v.z - mt2), p3 = __expf(v.w - mt2);
        sum += (p0 + p1) + (p2 + p3);
        const int cc = (c << 2) & 31;
        const int chunk = c >> 3;
        _Float16* dst = &Pf[(((strip << 1) + chunk) << 9) +
                            (a_lane(m, cc) << 4) + a_slot(cc)];
        store_h2(dst, p0, p1);
        store_h2(dst + 2, p2, p3);
      }
      mrow[tid] = mt2;
      lrow[tid] = lrow[tid] * sc + sum;
      scrow[tid] = sc;
    }
    __syncthreads();

    // Rescale accumulators, then O += P * V.
#pragma unroll
    for (int t = 0; t < 4; ++t)
#pragma unroll
      for (int r = 0; r < 8; ++r) acc[t][r] *= scrow[wave * 16 + r + 8 * hh];

    const v16h pa0 = load_frag(&Pf[((wave << 1) + 0) << 9], lane);
    const v16h pa1 = load_frag(&Pf[((wave << 1) + 1) << 9], lane);
#pragma unroll
    for (int t = 0; t < 4; ++t) {
      acc[t] = wmma16(pa0, load_frag(&Vf[((t << 1) + 0) << 9], lane), acc[t]);
      acc[t] = wmma16(pa1, load_frag(&Vf[((t << 1) + 1) << 9], lane), acc[t]);
    }
    __syncthreads();
  }

  // Finalize; store attn pre-swizzled (A-frag over rows m = node*16+g).
#pragma unroll
  for (int t = 0; t < 4; ++t) {
#pragma unroll
    for (int r = 0; r < 8; ++r) {
      const int row = wave * 16 + r + 8 * hh;
      const int m = (q0 + row) * N_GRAPH + g;
      const int e = head * HDIM + t * 16 + nc;
      const float inv = 1.0f / lrow[row];
      aws[a_ws_off(m, e)] = (_Float16)(acc[t][r] * inv);
    }
  }
}

// ---------------------------------------------------------------- kernel 3
// Output projection from pre-swizzled attn and wo.
__global__ __launch_bounds__(128) void out_proj_kernel(
    const _Float16* __restrict__ aws, const _Float16* __restrict__ wos,
    const float* __restrict__ bo, float* __restrict__ out) {
  __shared__ __align__(32) _Float16 Af[2048];
  __shared__ __align__(32) _Float16 Bf[2048];

  const int tid = threadIdx.x, lane = tid & 31, wave = tid >> 5;
  const int mt = blockIdx.x, nt = blockIdx.y;

  v8f acc[4];
#pragma unroll
  for (int t = 0; t < 4; ++t)
#pragma unroll
    for (int r = 0; r < 8; ++r) acc[t][r] = 0.f;

  for (int ch = 0; ch < NCH; ++ch) {
    const size_t abase = ((size_t)mt * NCH + ch) << 11;
    const size_t bbase = ((size_t)nt * NCH + ch) << 11;
#if HAS_TDM
    if (wave == 0) {
      tdm_load_lds(Af, aws + abase, 2048);
      tdm_load_lds(Bf, wos + bbase, 2048);
      __builtin_amdgcn_s_wait_tensorcnt(0);
    }
#else
    coop_copy(tid, 128, Af, aws + abase, 2048);
    coop_copy(tid, 128, Bf, wos + bbase, 2048);
#endif
    __syncthreads();
    const v16h a = load_frag(&Af[wave << 9], lane);
#pragma unroll
    for (int t = 0; t < 4; ++t)
      acc[t] = wmma16(a, load_frag(&Bf[t << 9], lane), acc[t]);
    __syncthreads();
  }

  const int hh = (lane >> 4) & 1, nc = lane & 15;
#pragma unroll
  for (int t = 0; t < 4; ++t) {
    const int n = nt * 64 + t * 16 + nc;
    const float bv = bo[n];
#pragma unroll
    for (int r = 0; r < 8; ++r) {
      const int m = mt * 64 + wave * 16 + r + 8 * hh;
      out[(size_t)m * EMBED + n] = acc[t][r] + bv;
    }
  }
}

// ---------------------------------------------------------------- launch
extern "C" void kernel_launch(void* const* d_in, const int* in_sizes, int n_in,
                              void* d_out, int out_size, void* d_ws, size_t ws_size,
                              hipStream_t stream) {
  (void)in_sizes; (void)n_in; (void)out_size; (void)ws_size;
  const float* x         = (const float*)d_in[0];
  const float* attn_bias = (const float*)d_in[1];
  const float* attn_mask = (const float*)d_in[2];
  const unsigned char* kpm = (const unsigned char*)d_in[3];
  const float* wq = (const float*)d_in[4];  const float* bq = (const float*)d_in[5];
  const float* wk = (const float*)d_in[6];  const float* bk = (const float*)d_in[7];
  const float* wv = (const float*)d_in[8];  const float* bv = (const float*)d_in[9];
  const float* wo = (const float*)d_in[10]; const float* bo = (const float*)d_in[11];
  float* out = (float*)d_out;

  char* ws = (char*)d_ws;
  const size_t mat   = (size_t)NROWS * EMBED * sizeof(_Float16);  // 12.6 MB
  const size_t wmatB = (size_t)EMBED * EMBED * sizeof(_Float16);  // 1.125 MB
  _Float16* xs  = (_Float16*)(ws);
  _Float16* qws = (_Float16*)(ws + mat);
  _Float16* kws = (_Float16*)(ws + 2 * mat);
  _Float16* vws = (_Float16*)(ws + 3 * mat);
  _Float16* aws = (_Float16*)(ws + 4 * mat);
  _Float16* wqs = (_Float16*)(ws + 5 * mat);
  _Float16* wks = (_Float16*)(ws + 5 * mat + wmatB);
  _Float16* wvs = (_Float16*)(ws + 5 * mat + 2 * wmatB);
  _Float16* wos = (_Float16*)(ws + 5 * mat + 3 * wmatB);

  xswz_kernel<<<dim3(NROWS), 256, 0, stream>>>(x, xs);
  wswz_kernel<<<dim3(EMBED, 4), 256, 0, stream>>>(wq, wk, wv, wo,
                                                  wqs, wks, wvs, wos);
  dim3 gproj(NROWS / 64, EMBED / 64);  // 128 x 12
  qkv_proj_kernel<<<gproj, 128, 0, stream>>>(xs, wqs, wks, wvs, bq, bk, bv,
                                             qws, kws, vws);
  dim3 gattn(N_NODE / 64, NB);         // 8 x 192
  attention_kernel<<<gattn, 128, 0, stream>>>(qws, kws, vws, attn_bias,
                                              attn_mask, kpm, aws);
  out_proj_kernel<<<gproj, 128, 0, stream>>>(aws, wos, bo, out);
}
